// Autoformer_21964462751922
// MI455X (gfx1250) — matrix operand order repaired and checked by reading, not compiled, not run on hardware
//
#include <hip/hip_runtime.h>

// Problem constants (must match reference)
#define Bz 8
#define Lz 2048
#define DM 1024
#define Hz 16
#define Ez 64
#define TOPK 7   // int(log(2048)) = 7

constexpr int GM = Bz * Lz;   // 16384 rows
constexpr int GN = DM;        // 1024 cols
constexpr int GK = DM;        // 1024 reduction

typedef __attribute__((ext_vector_type(16))) __bf16 v16bf;
typedef __attribute__((ext_vector_type(8)))  __bf16 v8bf;
typedef __attribute__((ext_vector_type(2)))  __bf16 v2bf;
typedef __attribute__((ext_vector_type(8)))  float  v8f;

static __device__ __forceinline__ __bf16 f2bf(float f) { return (__bf16)f; }

// ---------------------------------------------------------------------------
// 1) Column sum over L: out[b][c] = sum_l in[b][l][c]     (B x 1024)
// ---------------------------------------------------------------------------
__global__ void colsum_kernel(const float* __restrict__ in, float* __restrict__ out) {
    int b = blockIdx.x;
    int c = blockIdx.y * 128 + threadIdx.x;
    const float* p = in + (size_t)b * Lz * DM + c;
    float s = 0.0f;
    for (int l = 0; l < Lz; ++l) s += p[(size_t)l * DM];
    out[b * DM + c] = s;
}

// ---------------------------------------------------------------------------
// 2) S[b][n] = sum_k colsum[b][k] * W[k][n] + L*bias[n]   (8x1024 @ 1024x1024)
//    Full f32 — feeds the discrete top-k decision.
// ---------------------------------------------------------------------------
__global__ void smallproj_kernel(const float* __restrict__ cs_in,
                                 const float* __restrict__ W,
                                 const float* __restrict__ bias,
                                 float* __restrict__ out) {
    int n = blockIdx.x * 256 + threadIdx.x;
    int b = blockIdx.y;
    __shared__ float cs[DM];
    for (int i = threadIdx.x; i < DM; i += 256) cs[i] = cs_in[b * DM + i];
    __syncthreads();
    float s = 0.0f;
    for (int k = 0; k < DM; ++k) s += cs[k] * W[(size_t)k * DM + n];
    out[b * DM + n] = s + (float)Lz * bias[n];
}

// ---------------------------------------------------------------------------
// 3) Stats: mean_value[b][e] = (1/(H*L)) sum_h Sq[b,h,e]*Sk[b,h,e]  (B x 64)
//    batch-mean -> top-7 (first-occurrence tie break) -> per-batch softmax
// ---------------------------------------------------------------------------
__global__ void stats_kernel(const float* __restrict__ Sq, const float* __restrict__ Sk,
                             float* __restrict__ corrW, int* __restrict__ idxOut) {
    __shared__ float mv[Bz][Ez];
    __shared__ float meanB[Ez];
    __shared__ int   sidx[TOPK];
    int t = threadIdx.x;
    for (int i = t; i < Bz * Ez; i += 256) {
        int b = i >> 6, e = i & 63;
        float s = 0.0f;
        for (int h = 0; h < Hz; ++h)
            s += Sq[b * DM + h * Ez + e] * Sk[b * DM + h * Ez + e];
        mv[b][e] = s * (1.0f / ((float)Hz * (float)Lz));
    }
    __syncthreads();
    if (t < Ez) {
        float s = 0.0f;
        for (int b = 0; b < Bz; ++b) s += mv[b][t];
        meanB[t] = s * (1.0f / (float)Bz);
    }
    __syncthreads();
    if (t == 0) {
        for (int i = 0; i < TOPK; ++i) {
            int best = 0; float bv = -3.4e38f;
            for (int e = 0; e < Ez; ++e) {
                bool taken = false;
                for (int j = 0; j < i; ++j) if (sidx[j] == e) taken = true;
                if (!taken && meanB[e] > bv) { bv = meanB[e]; best = e; }
            }
            sidx[i] = best;
            idxOut[i] = best;
        }
    }
    __syncthreads();
    if (t < Bz) {
        float w[TOPK]; float m = -3.4e38f;
        for (int i = 0; i < TOPK; ++i) { w[i] = mv[t][sidx[i]]; m = fmaxf(m, w[i]); }
        float sum = 0.0f;
        for (int i = 0; i < TOPK; ++i) { w[i] = __expf(w[i] - m); sum += w[i]; }
        float inv = 1.0f / sum;
        for (int i = 0; i < TOPK; ++i) corrW[t * TOPK + i] = w[i] * inv;
    }
}

// ---------------------------------------------------------------------------
// 4) WMMA bf16 GEMM: C[16384,1024] = A(f32) * W(f32) + bias, f32 out.
//    Double-buffered LDS, one barrier per K-step; next tile's global loads
//    overlap current tile's WMMAs.
//    - A tile row-major bf16: fragment = 2x aligned ds_load_b128
//    - W tile TRANSPOSED (BsT[col][k]): fragment = 2x aligned ds_load_b128;
//      staged as K-pairs -> v_cvt_pk_bf16_f32 + ds_store_b32 per column.
//    Block 256 thr (8 waves); 64x64 tile; K-step 32; 2 WMMA tiles/wave.
// ---------------------------------------------------------------------------
__global__ __launch_bounds__(256) void gemm_bf16_kernel(
    const float* __restrict__ A, const float* __restrict__ W,
    const float* __restrict__ bias, float* __restrict__ C) {
    // stride 40 bf16 = 80 bytes (multiple of 16 -> aligned b128 fragment loads)
    __shared__ __attribute__((aligned(16))) __bf16 As [2][64][40];  // [buf][row][k]
    __shared__ __attribute__((aligned(16))) __bf16 BsT[2][64][40];  // [buf][col][k]

    const int tid  = threadIdx.x;
    const int bm   = blockIdx.x;
    const int bn   = blockIdx.y;
    const int wave = tid >> 5;
    const int lane = tid & 31;
    const int lr   = lane & 15;
    const int hi   = (lane >= 16) ? 1 : 0;
    const int tm   = wave & 3;            // 16-row tile within 64
    const int tn0  = (wave >> 2) * 32;    // two 16-col tiles

    v8f acc0 = {};
    v8f acc1 = {};

    // A loader: thread -> (row ar, k ac..ac+7)
    const int ar = tid >> 2;              // 0..63
    const int ac = (tid & 3) * 8;         // 0/8/16/24
    // B loader: thread -> (k rows 2*brr,2*brr+1, cols bcc..bcc+3)
    const int brr = tid >> 4;             // 0..15 -> k pair base
    const int bcc = (tid & 15) * 4;       // 0..60
    const float* ap  = A + ((size_t)bm * 64 + ar) * GK + ac;
    const float* wp0 = W + (size_t)(2 * brr) * GN + (size_t)bn * 64 + bcc;

    const int arow = tm * 16 + lr;
    const int kh   = hi ? 16 : 0;         // A: K half per lane group
    const int kb   = hi ? 16 : 0;         // B: K half per lane group

    auto stage = [&](int s, int k0) {
        // A 64x32 (f32 -> bf16, one b128 store)
        const float4 a0 = *(const float4*)(ap + k0);
        const float4 a1 = *(const float4*)(ap + k0 + 4);
        v8bf apk;
        apk[0] = f2bf(a0.x); apk[1] = f2bf(a0.y); apk[2] = f2bf(a0.z); apk[3] = f2bf(a0.w);
        apk[4] = f2bf(a1.x); apk[5] = f2bf(a1.y); apk[6] = f2bf(a1.z); apk[7] = f2bf(a1.w);
        *(v8bf*)&As[s][ar][ac] = apk;

        // W 32x64 transposed: K-adjacent pair per column -> b32 stores
        const float* wp = wp0 + (size_t)k0 * GN;
        const float4 wlo = *(const float4*)(wp);          // k = k0+2*brr
        const float4 whi = *(const float4*)(wp + GN);     // k = k0+2*brr+1
        v2bf p0; p0[0] = f2bf(wlo.x); p0[1] = f2bf(whi.x);
        v2bf p1; p1[0] = f2bf(wlo.y); p1[1] = f2bf(whi.y);
        v2bf p2; p2[0] = f2bf(wlo.z); p2[1] = f2bf(whi.z);
        v2bf p3; p3[0] = f2bf(wlo.w); p3[1] = f2bf(whi.w);
        *(v2bf*)&BsT[s][bcc + 0][2 * brr] = p0;
        *(v2bf*)&BsT[s][bcc + 1][2 * brr] = p1;
        *(v2bf*)&BsT[s][bcc + 2][2 * brr] = p2;
        *(v2bf*)&BsT[s][bcc + 3][2 * brr] = p3;
    };

    stage(0, 0);

    for (int k0 = 0; k0 < GK; k0 += 32) {
        const int s = (k0 >> 5) & 1;
        __syncthreads();                       // buffer s fully staged
        if (k0 + 32 < GK) stage(s ^ 1, k0 + 32);

        // A fragment: K kh..kh+15 contiguous, 2x b128
        const v8bf a_lo = *(const v8bf*)&As[s][arow][kh];
        const v8bf a_hi = *(const v8bf*)&As[s][arow][kh + 8];
        const v16bf af = __builtin_shufflevector(a_lo, a_hi,
            0, 1, 2, 3, 4, 5, 6, 7, 8, 9, 10, 11, 12, 13, 14, 15);

        // B fragments: K kb..kb+15 contiguous per column, 2x b128 each
        const v8bf b0lo = *(const v8bf*)&BsT[s][tn0 + lr][kb];
        const v8bf b0hi = *(const v8bf*)&BsT[s][tn0 + lr][kb + 8];
        const v16bf bf0 = __builtin_shufflevector(b0lo, b0hi,
            0, 1, 2, 3, 4, 5, 6, 7, 8, 9, 10, 11, 12, 13, 14, 15);
        const v8bf b1lo = *(const v8bf*)&BsT[s][tn0 + 16 + lr][kb];
        const v8bf b1hi = *(const v8bf*)&BsT[s][tn0 + 16 + lr][kb + 8];
        const v16bf bf1 = __builtin_shufflevector(b1lo, b1hi,
            0, 1, 2, 3, 4, 5, 6, 7, 8, 9, 10, 11, 12, 13, 14, 15);

        acc0 = __builtin_amdgcn_wmma_f32_16x16x32_bf16(false, af, false, bf0,
                                                       (short)0, acc0, false, false);
        acc1 = __builtin_amdgcn_wmma_f32_16x16x32_bf16(false, af, false, bf1,
                                                       (short)0, acc1, false, false);
        // note: no second barrier — double buffering; each wave's LDS reads
        // complete (s_wait_dscnt before wmma) before it reaches the next barrier
    }

    // C/D layout: vgpr v -> row v (lanes 0-15) / v+8 (lanes 16-31); col = lane&15
    const int rowb = bm * 64 + tm * 16 + hi * 8;
    const int col0 = bn * 64 + tn0 + lr;
    const float bia0 = bias[col0];
    const float bia1 = bias[col0 + 16];
    float* cp = C + (size_t)rowb * GN + col0;
    #pragma unroll
    for (int v = 0; v < 8; ++v) {
        cp[v * GN]      = acc0[v] + bia0;
        cp[v * GN + 16] = acc1[v] + bia1;
    }
}

// ---------------------------------------------------------------------------
// 5) agg[b,l,:] = sum_i corrW[b,i] * V[b, (l + idx[i]) % L, :]
// ---------------------------------------------------------------------------
__global__ void agg_kernel(const float* __restrict__ V, const int* __restrict__ idx,
                           const float* __restrict__ cw, float* __restrict__ agg) {
    const int row = blockIdx.x;
    const int b = row >> 11;       // L = 2048
    const int l = row & (Lz - 1);
    const int c = threadIdx.x * 4;
    float wloc[TOPK]; int iloc[TOPK];
    #pragma unroll
    for (int i = 0; i < TOPK; ++i) { wloc[i] = cw[b * TOPK + i]; iloc[i] = idx[i]; }
    float4 acc = {0.0f, 0.0f, 0.0f, 0.0f};
    #pragma unroll
    for (int i = 0; i < TOPK; ++i) {
        const int src = (l + iloc[i]) & (Lz - 1);
        const float4 vv = *(const float4*)(V + ((size_t)b * Lz + src) * DM + c);
        acc.x += wloc[i] * vv.x; acc.y += wloc[i] * vv.y;
        acc.z += wloc[i] * vv.z; acc.w += wloc[i] * vv.w;
    }
    *(float4*)(agg + (size_t)row * DM + c) = acc;
}

// ---------------------------------------------------------------------------
extern "C" void kernel_launch(void* const* d_in, const int* in_sizes, int n_in,
                              void* d_out, int out_size, void* d_ws, size_t ws_size,
                              hipStream_t stream) {
    (void)in_sizes; (void)n_in; (void)out_size; (void)ws_size;
    const float* queries = (const float*)d_in[0];
    const float* keys    = (const float*)d_in[1];
    const float* values  = (const float*)d_in[2];
    const float* Wq = (const float*)d_in[3];
    const float* bq = (const float*)d_in[4];
    const float* Wk = (const float*)d_in[5];
    const float* bk = (const float*)d_in[6];
    const float* Wv = (const float*)d_in[7];
    const float* bv = (const float*)d_in[8];
    const float* Wo = (const float*)d_in[9];
    const float* bo = (const float*)d_in[10];
    float* out = (float*)d_out;

    char* ws = (char*)d_ws;
    const size_t big = (size_t)Bz * Lz * DM * sizeof(float);   // 64 MB
    float* Vf    = (float*)(ws);
    float* aggf  = (float*)(ws + big);
    float* sqin  = (float*)(ws + 2 * big);
    float* skin  = sqin + Bz * DM;
    float* Sq    = skin + Bz * DM;
    float* Sk    = Sq + Bz * DM;
    float* corrW = Sk + Bz * DM;      // B*TOPK floats
    int*   idx   = (int*)(corrW + 64);

    // statistics path (exact f32)
    colsum_kernel<<<dim3(Bz, 8), 128, 0, stream>>>(queries, sqin);
    colsum_kernel<<<dim3(Bz, 8), 128, 0, stream>>>(keys, skin);
    smallproj_kernel<<<dim3(4, Bz), 256, 0, stream>>>(sqin, Wq, bq, Sq);
    smallproj_kernel<<<dim3(4, Bz), 256, 0, stream>>>(skin, Wk, bk, Sk);
    stats_kernel<<<1, 256, 0, stream>>>(Sq, Sk, corrW, idx);

    // heavy path (WMMA bf16, f32 accumulate)
    gemm_bf16_kernel<<<dim3(GM / 64, GN / 64), 256, 0, stream>>>(values, Wv, bv, Vf);
    agg_kernel<<<dim3(GM), 256, 0, stream>>>(Vf, idx, corrW, aggf);
    gemm_bf16_kernel<<<dim3(GM / 64, GN / 64), 256, 0, stream>>>(aggf, Wo, bo, out);
}
